// E8P12_codebook_86363202388343
// MI455X (gfx1250) — matrix operand order repaired and checked by reading, not compiled, not run on hardware
//
#include <hip/hip_runtime.h>

// ---- CDNA5 ext-vector types (POD, bit_cast/shuffle friendly) ----
typedef __attribute__((ext_vector_type(16))) __bf16   v16bf;
typedef __attribute__((ext_vector_type(8)))  __bf16   v8bf;
typedef __attribute__((ext_vector_type(4)))  __bf16   bf16x4;
typedef __attribute__((ext_vector_type(8)))  float    v8f;
typedef __attribute__((ext_vector_type(4)))  float    f32x4;
typedef __attribute__((ext_vector_type(4)))  unsigned u32x4;
typedef __attribute__((ext_vector_type(2)))  int      i32x2;

// Fixed problem shape (reference setup_inputs)
#define B_TOK 4096   // tokens
#define M_OUT 4096   // out features == Qidxs rows
#define N_COD 512    // codes per row
#define K_IN  4096   // in features == N_COD*8
#define GRID_ROWS 65536

// Block tiling: 256 threads = 8 waves. Output tile 128(tokens) x 128(features).
// Wave w owns feature columns [mBase + 16w, +16) across all 128 token rows, so
// each wave does ONE codebook gather per 32-K chunk, reused by 8 WMMAs.
#define TILE_B 128
#define TILE_M 128
#define TILE_K 32
#define NCHUNK (K_IN / TILE_K)

// ---------------------------------------------------------------------------
// Prologue kernel: convert the f32 codebook [65536, 8] to bf16 rows (exact:
// all entries are multiples of 0.25 with |w| <= 3.75). One 16B row per code
// -> the GEMM gathers ONE b128 per code and skips per-use converts.
// ---------------------------------------------------------------------------
__global__ __launch_bounds__(256) void grid_to_bf16(
    const float* __restrict__ grid, u32x4* __restrict__ gridb)
{
    const int i = blockIdx.x * 256 + threadIdx.x;      // codebook row
    const f32x4* s = (const f32x4*)grid + (size_t)i * 2;
    f32x4 a = s[0], b = s[1];
    bf16x4 c0 = __builtin_convertvector(a, bf16x4);
    bf16x4 c1 = __builtin_convertvector(b, bf16x4);
    v8bf v = __builtin_shufflevector(c0, c1, 0,1,2,3,4,5,6,7);
    gridb[i] = __builtin_bit_cast(u32x4, v);
}

// ---------------------------------------------------------------------------
// Main GEMM: C[b,m] = sum_k input[b,k] * grid[Qidxs[m, k/8], k%8]
// Software-pipelined 2 deep: while chunk kc runs its 8 WMMAs, the codebook
// gathers + A-tile global loads for kc+1 are in flight (and the Qidxs codes
// for kc+2), so no load wait sits on the critical path.
// TBL=true gathers the prebuilt bf16 table in d_ws; TBL=false gathers f32.
// ---------------------------------------------------------------------------
template <bool TBL>
__global__ __launch_bounds__(256) void e8p_wmma_gemm(
    const float* __restrict__ input,   // [B_TOK, K_IN] f32 row-major
    const int*   __restrict__ Qidxs,   // [M_OUT, N_COD] int32 (codes 0..65535)
    const float* __restrict__ grid,    // [65536, 8] f32 codebook
    const u32x4* __restrict__ gridb,   // [65536] 16B bf16 codebook rows (d_ws)
    float*       __restrict__ out)     // [B_TOK, M_OUT] f32
{
    // A-chunk in WMMA-A fragment layout:
    //   8 fragments (16 token rows each) x 32 lane-slots x 32 bytes = 8 KB.
    //   slot(f, lane): bytes[0:16) = row 16f+(lane&15), K = (lane>>4)*16 + 0..7
    //                  bytes[16:32) = same row,          K = (lane>>4)*16 + 8..15
    __shared__ u32x4 ldsA[(TILE_B / 16) * 32 * 2];

    const int t    = threadIdx.x;
    const int lane = t & 31;
    const int w    = t >> 5;                 // wave id 0..7 -> feature column strip
    const int bBase = blockIdx.y * TILE_B;
    const int mBase = blockIdx.x * TILE_M;

    // ---- staging coords: thread t converts row (t>>1), 16-wide k-half (t&1) ----
    const int sr  = t >> 1;                  // 0..127 token row within tile
    const int kh  = t & 1;                   // which 16-k half of the 32-k chunk
    const int sf  = sr >> 4;                 // A fragment index
    const int sri = sr & 15;                 // row within fragment
    const float* srcRow = input + (size_t)(bBase + sr) * K_IN + kh * 16;

    // ---- B-fragment coords (bf16 B 32x16, wave32, k order chosen here to match the A slots):
    //   lane L: column N = L&15; K = (L>>4)*16 + 0..15  == two consecutive codes
    const int  mcol = mBase + w * 16 + (lane & 15);
    const int  khB  = lane >> 4;
    const int* qrow = Qidxs + (size_t)mcol * N_COD;
    const f32x4* grid4 = (const f32x4*)grid; // 2 x f32x4 per code (fallback path)

    v8f acc[8] = {};                         // 8 x (16x16 f32) = 64 VGPRs

    // ---------------- pipeline prologue ----------------
    // codes for chunk 0, gather chunk 0, codes for chunk 1, A-tile chunk 0
    i32x2 qA = *(const i32x2*)(qrow + khB * 2);
    u32x4 r0, r1;
    f32x4 g0a, g0b, g1a, g1b;
    if constexpr (TBL) {
        r0 = gridb[(size_t)(unsigned)qA.x];
        r1 = gridb[(size_t)(unsigned)qA.y];
    } else {
        g0a = grid4[(size_t)(unsigned)qA.x * 2 + 0];
        g0b = grid4[(size_t)(unsigned)qA.x * 2 + 1];
        g1a = grid4[(size_t)(unsigned)qA.y * 2 + 0];
        g1b = grid4[(size_t)(unsigned)qA.y * 2 + 1];
    }
    i32x2 qB = *(const i32x2*)(qrow + 4 + khB * 2);
    f32x4 sa0, sa1, sa2, sa3;
    {
        const f32x4* s = (const f32x4*)srcRow;
        sa0 = s[0]; sa1 = s[1]; sa2 = s[2]; sa3 = s[3];
    }

    for (int kc = 0; kc < NCHUNK; ++kc) {
        // ---------- stage A chunk f32 -> bf16 into LDS (fragment layout) ----
        __syncthreads();                     // previous iter's LDS reads done
        {
            bf16x4 c0 = __builtin_convertvector(sa0, bf16x4);
            bf16x4 c1 = __builtin_convertvector(sa1, bf16x4);
            bf16x4 c2 = __builtin_convertvector(sa2, bf16x4);
            bf16x4 c3 = __builtin_convertvector(sa3, bf16x4);
            v8bf lo = __builtin_shufflevector(c0, c1, 0,1,2,3,4,5,6,7); // K 16kh+0..7
            v8bf hi = __builtin_shufflevector(c2, c3, 0,1,2,3,4,5,6,7); // K 16kh+8..15
            ldsA[(sf * 32 + kh * 16 + sri) * 2 + 0] = __builtin_bit_cast(u32x4, lo);
            ldsA[(sf * 32 + kh * 16 + sri) * 2 + 1] = __builtin_bit_cast(u32x4, hi);
        }

        // ---------- build B fragment for THIS chunk (payload gathered earlier)
        v16bf bfrag;
        if constexpr (TBL) {
            v8bf blo = __builtin_bit_cast(v8bf, r0);
            v8bf bhi = __builtin_bit_cast(v8bf, r1);
            bfrag = __builtin_shufflevector(blo, bhi,
                        0,1,2,3,4,5,6,7,8,9,10,11,12,13,14,15);
        } else {
            bf16x4 b0 = __builtin_convertvector(g0a, bf16x4);
            bf16x4 b1 = __builtin_convertvector(g0b, bf16x4);
            bf16x4 b2 = __builtin_convertvector(g1a, bf16x4);
            bf16x4 b3 = __builtin_convertvector(g1b, bf16x4);
            v8bf blo = __builtin_shufflevector(b0, b1, 0,1,2,3,4,5,6,7);
            v8bf bhi = __builtin_shufflevector(b2, b3, 0,1,2,3,4,5,6,7);
            bfrag = __builtin_shufflevector(blo, bhi,
                        0,1,2,3,4,5,6,7,8,9,10,11,12,13,14,15);
        }

        // ---------- issue NEXT chunk's loads (covered by the WMMA section) --
        if (kc + 1 < NCHUNK) {
            if constexpr (TBL) {                       // gather kc+1 (codes ready)
                r0 = gridb[(size_t)(unsigned)qB.x];
                r1 = gridb[(size_t)(unsigned)qB.y];
            } else {
                g0a = grid4[(size_t)(unsigned)qB.x * 2 + 0];
                g0b = grid4[(size_t)(unsigned)qB.x * 2 + 1];
                g1a = grid4[(size_t)(unsigned)qB.y * 2 + 0];
                g1b = grid4[(size_t)(unsigned)qB.y * 2 + 1];
            }
            const f32x4* s = (const f32x4*)(srcRow + (size_t)(kc + 1) * TILE_K);
            sa0 = s[0]; sa1 = s[1]; sa2 = s[2]; sa3 = s[3];
            if (kc + 2 < NCHUNK)                       // codes for kc+2
                qB = *(const i32x2*)(qrow + (kc + 2) * 4 + khB * 2);
        }
        __syncthreads();

        // ---------- 8 WMMAs, A fragments double-buffered one ahead ----------
        u32x4 alo = ldsA[(0 * 32 + lane) * 2 + 0];
        u32x4 ahi = ldsA[(0 * 32 + lane) * 2 + 1];
        #pragma unroll
        for (int f = 0; f < 8; ++f) {
            u32x4 nlo, nhi;
            if (f < 7) {
                nlo = ldsA[((f + 1) * 32 + lane) * 2 + 0];
                nhi = ldsA[((f + 1) * 32 + lane) * 2 + 1];
            }
            v8bf  a0 = __builtin_bit_cast(v8bf, alo);
            v8bf  a1 = __builtin_bit_cast(v8bf, ahi);
            v16bf afrag = __builtin_shufflevector(a0, a1,
                            0,1,2,3,4,5,6,7,8,9,10,11,12,13,14,15);
            acc[f] = __builtin_amdgcn_wmma_f32_16x16x32_bf16(
                        /*neg_a=*/false, afrag,
                        /*neg_b=*/false, bfrag,
                        /*c_mod=*/(short)0, acc[f],
                        /*reuse_a=*/false, /*reuse_b=*/false);
            alo = nlo; ahi = nhi;
        }
    }

    // ---------- store: D 16x16 f32 layout: lane L col N=L&15; VGPR r -> row r + 8*(L>>4)
    const int ocol  = mBase + w * 16 + (lane & 15);
    const int rhalf = (lane >> 4) * 8;
    #pragma unroll
    for (int f = 0; f < 8; ++f) {
        #pragma unroll
        for (int r = 0; r < 8; ++r) {
            const int orow = bBase + f * 16 + rhalf + r;
            out[(size_t)orow * M_OUT + ocol] = acc[f][r];
        }
    }
}

extern "C" void kernel_launch(void* const* d_in, const int* in_sizes, int n_in,
                              void* d_out, int out_size, void* d_ws, size_t ws_size,
                              hipStream_t stream) {
    const float* input = (const float*)d_in[0];   // [4096, 4096] f32
    const int*   Qidxs = (const int*)d_in[1];     // [4096, 512] i32
    const float* grid  = (const float*)d_in[2];   // [65536, 8] f32
    float*       out   = (float*)d_out;           // [4096, 4096] f32

    dim3 grd(M_OUT / TILE_M, B_TOK / TILE_B);     // 32 x 32 blocks

    const size_t tblBytes = (size_t)GRID_ROWS * 16; // 1 MB bf16 codebook
    if (ws_size >= tblBytes) {
        u32x4* gridb = (u32x4*)d_ws;
        grid_to_bf16<<<GRID_ROWS / 256, 256, 0, stream>>>(grid, gridb);
        e8p_wmma_gemm<true><<<grd, 256, 0, stream>>>(input, Qidxs, grid, gridb, out);
    } else {
        e8p_wmma_gemm<false><<<grd, 256, 0, stream>>>(input, Qidxs, grid, nullptr, out);
    }
}
